// TFLSHAttention_11905649344820
// MI455X (gfx1250) — matrix operand order repaired and checked by reading, not compiled, not run on hardware
//
#include <hip/hip_runtime.h>
#include <math.h>

// ---- CDNA5 vector types -----------------------------------------------------
typedef __attribute__((ext_vector_type(16))) _Float16 v16h;
typedef __attribute__((ext_vector_type(8)))  _Float16 v8h;
typedef __attribute__((ext_vector_type(8)))  float    v8f;

#define LSH_B 16
#define LSH_S 4096
#define LSH_D 64
#define LSH_H 8
#define LSH_NB 64          // buckets per hash round
#define LSH_C  512         // global chunks = H * (S/64)
#define LSH_BK 64          // bucket/chunk size

static __device__ __forceinline__ void atomic_add_f32(float* p, float x) {
  __hip_atomic_fetch_add(p, x, __ATOMIC_RELAXED, __HIP_MEMORY_SCOPE_AGENT);
}

// ---- kernel 0: zero the accumulators (num + den are contiguous) -------------
__global__ void lsh_zero_kernel(float* __restrict__ p, int n) {
  int i = blockIdx.x * 256 + threadIdx.x;
  if (i < n) p[i] = 0.0f;
}

// ---- kernel 1: hashing ------------------------------------------------------
// rotated[b,h,t,i] = sum_f qk[b,t,f] * rot[f,h,i];  bucket = argmax([r,-r])
__global__ __launch_bounds__(256) void lsh_hash_kernel(
    const float* __restrict__ qk, const float* __restrict__ rot,
    int* __restrict__ bucket_ws, int* __restrict__ buckets_out) {
  __shared__ __align__(16) float rot_s[64 * 256];   // [f][h*32+i], 64 KB
  const int tid = threadIdx.x;
  for (int k = tid; k < 64 * 256; k += 256) rot_s[k] = rot[k];
  __syncthreads();

  const int idx = blockIdx.x * 256 + tid;           // b*S + t
  const int b = idx >> 12;
  const int t = idx & 4095;

  float q[64];
  const float4* src = (const float4*)(qk + (size_t)idx * 64);
#pragma unroll
  for (int d4 = 0; d4 < 16; ++d4) {
    float4 x = src[d4];
    q[4 * d4 + 0] = x.x; q[4 * d4 + 1] = x.y;
    q[4 * d4 + 2] = x.z; q[4 * d4 + 3] = x.w;
  }

  for (int h = 0; h < LSH_H; ++h) {
    float bestmax = -3.4e38f, bestmin = 3.4e38f;
    int imax = 0, imin = 0;
    for (int ib = 0; ib < 4; ++ib) {                // 8 rotation outputs per block
      float acc[8] = {0, 0, 0, 0, 0, 0, 0, 0};
#pragma unroll
      for (int f = 0; f < 64; ++f) {
        const float* rp = &rot_s[f * 256 + h * 32 + ib * 8];
        float4 r0 = *(const float4*)rp;
        float4 r1 = *(const float4*)(rp + 4);
        float qf = q[f];
        acc[0] = fmaf(qf, r0.x, acc[0]); acc[1] = fmaf(qf, r0.y, acc[1]);
        acc[2] = fmaf(qf, r0.z, acc[2]); acc[3] = fmaf(qf, r0.w, acc[3]);
        acc[4] = fmaf(qf, r1.x, acc[4]); acc[5] = fmaf(qf, r1.y, acc[5]);
        acc[6] = fmaf(qf, r1.z, acc[6]); acc[7] = fmaf(qf, r1.w, acc[7]);
      }
#pragma unroll
      for (int k = 0; k < 8; ++k) {
        int i = ib * 8 + k;
        if (acc[k] > bestmax) { bestmax = acc[k]; imax = i; }  // first max wins
        if (acc[k] < bestmin) { bestmin = acc[k]; imin = i; }  // first min wins
      }
    }
    // argmax over concat([r, -r]): ties prefer the r side (lower index)
    int bucket = (bestmax >= -bestmin) ? imax : (32 + imin);
    bucket_ws[(b * LSH_H + h) * LSH_S + t] = bucket;
    buckets_out[b * (LSH_H * LSH_S) + h * LSH_S + t] = bucket + h * LSH_NB;
  }
}

// ---- kernel 2: exact stable counting sort per (b,h) -------------------------
// st[b, h*S + rank] = original t, sorted by (bucket, t) ascending.
__global__ __launch_bounds__(64) void lsh_sort_kernel(
    const int* __restrict__ bucket_ws, int* __restrict__ st) {
  __shared__ int sb[LSH_S];
  __shared__ int cnt[64];
  __shared__ int csum[64];
  const int tid = threadIdx.x;          // thread == bucket id
  const int bh = blockIdx.x;            // b*H + h
  const int* src = bucket_ws + bh * LSH_S;
  for (int k = tid; k < LSH_S; k += 64) sb[k] = src[k];
  __syncthreads();
  int c = 0;
  for (int t = 0; t < LSH_S; ++t) c += (sb[t] == tid) ? 1 : 0;  // LDS broadcast
  cnt[tid] = c;
  __syncthreads();
  if (tid == 0) {
    int run = 0;
    for (int j = 0; j < 64; ++j) { csum[j] = run; run += cnt[j]; }
  }
  __syncthreads();
  int w = bh * LSH_S + csum[tid];
  for (int t = 0; t < LSH_S; ++t)
    if (sb[t] == tid) st[w++] = t;      // ascending t => stable
}

// ---- kernel 3: chunked attention with WMMA ----------------------------------
// One block = one (b, global chunk g). 128 threads = 4 wave32s; wave w owns
// query rows [16w, 16w+16). M=64, N=128 (chunk + look-back), K=D=64.
__global__ __launch_bounds__(128) void lsh_attn_kernel(
    const float* __restrict__ qk, const float* __restrict__ v,
    const int* __restrict__ st, float* __restrict__ num, float* __restrict__ den) {
  __shared__ int tk_s[128];                         // token ids of the 128 keys
  __shared__ __align__(16) _Float16 Qh[64 * 72];    // queries, row-major, pad 72
  __shared__ __align__(16) _Float16 Kh[128 * 72];   // normalized keys, row-major
  __shared__ __align__(16) _Float16 Vt[64 * 136];   // V transposed: [dim][key]
  __shared__ __align__(16) _Float16 Ph[4 * 16 * 136]; // per-wave probs [row][key]
  __shared__ float scale_s[64];                     // exp(rowmax) per query row

  const int tid = threadIdx.x;
  const int b = blockIdx.x >> 9;
  const int g = blockIdx.x & 511;

  // ---- stage: gather Q/K/V through the sort permutation ----
  const int chunk = (tid < 64) ? g : ((g + LSH_C - 1) & (LSH_C - 1)); // look-back wraps
  const int jj = tid & 63;
  const int tk = st[b * (LSH_H * LSH_S) + chunk * LSH_BK + jj];
  tk_s[tid] = tk;
  {
    const float4* src = (const float4*)(qk + ((size_t)(b * LSH_S + tk)) * 64);
    float f[64];
    float ss = 0.0f;
#pragma unroll
    for (int d4 = 0; d4 < 16; ++d4) {
      float4 x = src[d4];
      f[4 * d4 + 0] = x.x; f[4 * d4 + 1] = x.y;
      f[4 * d4 + 2] = x.z; f[4 * d4 + 3] = x.w;
      ss = fmaf(x.x, x.x, ss); ss = fmaf(x.y, x.y, ss);
      ss = fmaf(x.z, x.z, ss); ss = fmaf(x.w, x.w, ss);
    }
    const float nsc = 1.0f / (sqrtf(ss) + 1e-6f);   // bk = bqk / (|bqk| + 1e-6)
    // packed b128 stores: 8 halves per ds_store
#pragma unroll
    for (int d8 = 0; d8 < 8; ++d8) {
      v8h pk;
#pragma unroll
      for (int e = 0; e < 8; ++e) pk[e] = (_Float16)(f[d8 * 8 + e] * nsc);
      *(v8h*)(&Kh[tid * 72 + d8 * 8]) = pk;
    }
    if (tid < 64) {
#pragma unroll
      for (int d8 = 0; d8 < 8; ++d8) {
        v8h pq;
#pragma unroll
        for (int e = 0; e < 8; ++e) pq[e] = (_Float16)f[d8 * 8 + e];
        *(v8h*)(&Qh[tid * 72 + d8 * 8]) = pq;
      }
    }
  }
  {
    const float4* srcv = (const float4*)(v + ((size_t)(b * LSH_S + tk)) * 64);
#pragma unroll
    for (int d4 = 0; d4 < 16; ++d4) {
      float4 x = srcv[d4];
      Vt[(4 * d4 + 0) * 136 + tid] = (_Float16)x.x;
      Vt[(4 * d4 + 1) * 136 + tid] = (_Float16)x.y;
      Vt[(4 * d4 + 2) * 136 + tid] = (_Float16)x.z;
      Vt[(4 * d4 + 3) * 136 + tid] = (_Float16)x.w;
    }
  }
  __syncthreads();

  const int lane = tid & 31;
  const int wave = tid >> 5;
  const int qrow0 = wave * 16;
  const int lm = lane & 15;       // A row / B col / D col
  const int lh = lane >> 4;       // which 16-lane half
  const int kk0 = lh * 16;        // A-frag K sub-offset (halves)
  const int kb = lh * 16;         // B-frag K sub-offset (halves)
  const int rbase = lh * 8;       // D rows base (this lane holds rows rbase..rbase+7)

  // ---- GEMM 1: dots = Q (64x64) x K^T -> strip 16x128 ----
  v8f acc[8];
#pragma unroll
  for (int nt = 0; nt < 8; ++nt) acc[nt] = (v8f){};
#pragma unroll
  for (int ks = 0; ks < 2; ++ks) {
    const _Float16* arow = &Qh[(qrow0 + lm) * 72 + ks * 32];
    v8h alo = *(const v8h*)(arow + kk0);
    v8h ahi = *(const v8h*)(arow + kk0 + 8);
    v16h af = __builtin_shufflevector(alo, ahi, 0, 1, 2, 3, 4, 5, 6, 7,
                                      8, 9, 10, 11, 12, 13, 14, 15);
#pragma unroll
    for (int nt = 0; nt < 8; ++nt) {
      const _Float16* brow = &Kh[(nt * 16 + lm) * 72 + ks * 32 + kb];
      v8h blo = *(const v8h*)brow;
      v8h bhi = *(const v8h*)(brow + 8);
      v16h bf = __builtin_shufflevector(blo, bhi, 0, 1, 2, 3, 4, 5, 6, 7,
                                        8, 9, 10, 11, 12, 13, 14, 15);
      acc[nt] = __builtin_amdgcn_wmma_f32_16x16x32_f16(
          false, af, false, bf, (short)0, acc[nt], false, false);
    }
  }

  // ---- mask (keep only matching token positions) + scaled softmax ----
  int tq[8];
#pragma unroll
  for (int r = 0; r < 8; ++r) tq[r] = tk_s[qrow0 + rbase + r];
#pragma unroll
  for (int nt = 0; nt < 8; ++nt) {
    const int tkc = tk_s[nt * 16 + lm];
#pragma unroll
    for (int r = 0; r < 8; ++r) {
      float val = acc[nt][r] * 0.125f;              // * D^-0.5
      acc[nt][r] = (tq[r] == tkc) ? val : -1.0e5f;
    }
  }
  float m[8], s[8];
#pragma unroll
  for (int r = 0; r < 8; ++r) {
    float mv = acc[0][r];
#pragma unroll
    for (int nt = 1; nt < 8; ++nt) mv = fmaxf(mv, acc[nt][r]);
    mv = fmaxf(mv, __shfl_xor(mv, 1, 32));
    mv = fmaxf(mv, __shfl_xor(mv, 2, 32));
    mv = fmaxf(mv, __shfl_xor(mv, 4, 32));
    mv = fmaxf(mv, __shfl_xor(mv, 8, 32));          // stays inside 16-lane half
    m[r] = mv;
    s[r] = 0.0f;
  }
#pragma unroll
  for (int nt = 0; nt < 8; ++nt) {
#pragma unroll
    for (int r = 0; r < 8; ++r) {
      float e = __expf(acc[nt][r] - m[r]);          // masked -> exp(-1e5-m) == 0
      acc[nt][r] = e;
      s[r] += e;
    }
  }
#pragma unroll
  for (int r = 0; r < 8; ++r) {
    float sv = s[r];
    sv += __shfl_xor(sv, 1, 32);
    sv += __shfl_xor(sv, 2, 32);
    sv += __shfl_xor(sv, 4, 32);
    sv += __shfl_xor(sv, 8, 32);
    s[r] = sv;
  }
  if (lm == 0) {  // lanes 0 and 16: one writer per query row
#pragma unroll
    for (int r = 0; r < 8; ++r) {
      const int row = qrow0 + rbase + r;
      const float em = __expf(m[r]);
      scale_s[row] = em;
      atomic_add_f32(&den[b * LSH_S + tq[r]], em * s[r]);  // exp(lse)
    }
  }
  // stash unnormalized probs for GEMM 2
#pragma unroll
  for (int nt = 0; nt < 8; ++nt) {
#pragma unroll
    for (int r = 0; r < 8; ++r)
      Ph[wave * (16 * 136) + (rbase + r) * 136 + nt * 16 + lm] =
          (_Float16)acc[nt][r];
  }
  __syncthreads();

  // ---- GEMM 2: U = exp(dots-m) (16x128) x V (128x64), scatter-accumulate ----
  const _Float16* pbase = &Ph[wave * (16 * 136)];
#pragma unroll
  for (int nt2 = 0; nt2 < 4; ++nt2) {
    v8f acc2 = (v8f){};
#pragma unroll
    for (int ks = 0; ks < 4; ++ks) {
      const _Float16* arow = pbase + lm * 136 + ks * 32;
      v8h alo = *(const v8h*)(arow + kk0);
      v8h ahi = *(const v8h*)(arow + kk0 + 8);
      v16h af = __builtin_shufflevector(alo, ahi, 0, 1, 2, 3, 4, 5, 6, 7,
                                        8, 9, 10, 11, 12, 13, 14, 15);
      const _Float16* brow = &Vt[(nt2 * 16 + lm) * 136 + ks * 32 + kb];
      v8h blo = *(const v8h*)brow;
      v8h bhi = *(const v8h*)(brow + 8);
      v16h bf = __builtin_shufflevector(blo, bhi, 0, 1, 2, 3, 4, 5, 6, 7,
                                        8, 9, 10, 11, 12, 13, 14, 15);
      acc2 = __builtin_amdgcn_wmma_f32_16x16x32_f16(
          false, af, false, bf, (short)0, acc2, false, false);
    }
#pragma unroll
    for (int r = 0; r < 8; ++r) {
      const int row = qrow0 + rbase + r;
      const int t = tk_s[row];
      const int dd = nt2 * 16 + lm;
      atomic_add_f32(&num[((size_t)(b * LSH_S + t)) * 64 + dd],
                     acc2[r] * scale_s[row]);       // o * exp(lse)
    }
  }
}

// ---- kernel 4: out = num / den ----------------------------------------------
__global__ void lsh_combine_kernel(const float* __restrict__ num,
                                   const float* __restrict__ den,
                                   float* __restrict__ out) {
  int i = blockIdx.x * 256 + threadIdx.x;
  if (i < LSH_B * LSH_S * LSH_D) out[i] = num[i] / den[i >> 6];
}

// ---- launcher ---------------------------------------------------------------
extern "C" void kernel_launch(void* const* d_in, const int* in_sizes, int n_in,
                              void* d_out, int out_size, void* d_ws, size_t ws_size,
                              hipStream_t stream) {
  (void)in_sizes; (void)n_in; (void)out_size; (void)ws_size;
  const float* qk  = (const float*)d_in[0];
  const float* v   = (const float*)d_in[1];
  const float* rot = (const float*)d_in[2];

  float* out = (float*)d_out;                               // [B,S,D] f32
  int* buckets_out = (int*)d_out + LSH_B * LSH_S * LSH_D;   // [B,H*S] i32 bits

  char* ws = (char*)d_ws;
  int*   bucket_ws = (int*)ws;                              // [B,H,S]  2 MB
  int*   st        = (int*)(ws + (2u << 20));               // [B,H,S]  2 MB
  float* num       = (float*)(ws + (4u << 20));             // [B,S,D] 16 MB
  float* den       = (float*)(ws + (4u << 20) + (16u << 20)); // [B,S] 256 KB

  const int nzero = LSH_B * LSH_S * LSH_D + LSH_B * LSH_S;  // num+den contiguous
  lsh_zero_kernel<<<(nzero + 255) / 256, 256, 0, stream>>>(num, nzero);

  lsh_hash_kernel<<<(LSH_B * LSH_S) / 256, 256, 0, stream>>>(
      qk, rot, bucket_ws, buckets_out);

  lsh_sort_kernel<<<LSH_B * LSH_H, 64, 0, stream>>>(bucket_ws, st);

  lsh_attn_kernel<<<LSH_B * LSH_C, 128, 0, stream>>>(qk, v, st, num, den);

  lsh_combine_kernel<<<(LSH_B * LSH_S * LSH_D + 255) / 256, 256, 0, stream>>>(
      num, den, out);
}
